// MPNEncoder_61125974557339
// MI455X (gfx1250) — compile-verified, matrix operand order checked by reading, not run on hardware
//
#include <hip/hip_runtime.h>
#include <hip/hip_bf16.h>

#define N_ATOMS   100000
#define N_BONDS   400000
#define ATOM_FDIM 133
#define BOND_FDIM 147
#define HIDDEN    300
#define DEPTH     6
#define HP        320            // padded hidden stride (20 x 16)
#define NBP       400064         // bond rows padded to 64 (6251 blocks)
#define NAP       100096         // atom rows padded to 64 (1564 blocks)

typedef __attribute__((ext_vector_type(16))) __bf16       bf16x16;
typedef __attribute__((ext_vector_type(8)))  __bf16       bf16x8;
typedef __attribute__((ext_vector_type(8)))  float        f32x8;
typedef __attribute__((ext_vector_type(4)))  unsigned int u32x4;

#define DEV __device__ __forceinline__

// scalar f32 -> bf16 (RNE) for cold paths
DEV unsigned short cvt_bf16(float f) {
    unsigned u = __builtin_bit_cast(unsigned, f);
    u += 0x7FFFu + ((u >> 16) & 1u);
    return (unsigned short)(u >> 16);
}
DEV float bf2f(unsigned short s) { return __builtin_bit_cast(float, (unsigned)s << 16); }

DEV bf16x8 load_b8(const unsigned short* p) {
    return __builtin_bit_cast(bf16x8, *(const u32x4*)p);
}
DEV void store_b8(unsigned short* p, bf16x8 v) {
    *(u32x4*)p = __builtin_bit_cast(u32x4, v);
}

// Convert 8 f32 results (rows rowBase..rowBase+7, one column) to bf16 and
// scatter-store.  Goes through u32 lanes only: element 2j = low half of p[j],
// element 2j+1 = high half (little-endian), avoiding __bf16 element extracts.
DEV void store_col8_bf16(unsigned short* __restrict__ buf, long rowBase, int col,
                         f32x8 v) {
    u32x4 p = __builtin_bit_cast(u32x4, __builtin_convertvector(v, bf16x8));
#pragma unroll
    for (int j = 0; j < 4; ++j) {
        buf[(rowBase + 2 * j)     * HP + col] = (unsigned short)(p[j] & 0xFFFFu);
        buf[(rowBase + 2 * j + 1) * HP + col] = (unsigned short)(p[j] >> 16);
    }
}

// Load one 16x32 bf16 fragment (A layout; B uses the same per-lane layout on
// W rows since B must be column-major = W^T).
DEV bf16x16 load_frag(const unsigned short* tile, int ldk, int lane) {
    const int l = lane & 15, half = lane >> 4;
    const unsigned short* p = tile + l * ldk + half * 8;
    u32x4 lo4 = *(const u32x4*)p;         // K = base + {0..7 | 8..15}
    u32x4 hi4 = *(const u32x4*)(p + 16);  // K = base + {16..23 | 24..31}
    bf16x8 lo = __builtin_bit_cast(bf16x8, lo4);
    bf16x8 hi = __builtin_bit_cast(bf16x8, hi4);
    bf16x16 a;
#pragma unroll
    for (int i = 0; i < 8; ++i) { a[i] = lo[i]; a[8 + i] = hi[i]; }
    return a;
}

DEV f32x8 wmma_bf16(bf16x16 a, bf16x16 b, f32x8 c) {
    return __builtin_amdgcn_wmma_f32_16x16x32_bf16(false, a, false, b,
                                                   (short)0, c, false, false);
}

// ---------------------------------------------------------------- weights
__global__ void k_convert_pad(const float* __restrict__ src,
                              unsigned short* __restrict__ dst,
                              int rows, int cols, int prows, int pcols) {
    int tid = blockIdx.x * blockDim.x + threadIdx.x;
    int total = prows * pcols;
    if (tid >= total) return;
    int r = tid / pcols, c = tid - r * pcols;
    float v = (r < rows && c < cols) ? src[(long)r * cols + c] : 0.f;
    dst[tid] = cvt_bf16(v);
}

// ------------------------------------------- b_input = f_bonds @ W_i^T, relu
__global__ __launch_bounds__(256)
void k_bond_input(const float* __restrict__ f_bonds,
                  const unsigned short* __restrict__ Wi,   // [320][160] bf16
                  unsigned short* __restrict__ b_input,    // [NBP][HP] bf16
                  unsigned short* __restrict__ b_msg) {    // [NBP][HP] bf16
    constexpr int KP = 160, KC = 5, NT = 20, MT = 64;
    extern __shared__ unsigned short As[];                 // [MT][KP]
    const int rowBase = blockIdx.x * MT;
    {   // stage A: 4 threads per row, 40 cols each, vector f32->bf16, zero-pad
        int r  = threadIdx.x >> 2;
        int c0 = (threadIdx.x & 3) * (KP / 4);
        int row = rowBase + r;
        bool valid = row <= N_BONDS;
        unsigned short* dst = As + r * KP + c0;
        const float* srcr = f_bonds + (long)row * BOND_FDIM;
#pragma unroll
        for (int g = 0; g < (KP / 4) / 8; ++g) {           // 5 x 8 cols
            f32x8 f;
#pragma unroll
            for (int i = 0; i < 8; ++i) {
                int c = c0 + g * 8 + i;
                f[i] = (valid && c < BOND_FDIM) ? srcr[c] : 0.f;
            }
            store_b8(dst + g * 8, __builtin_convertvector(f, bf16x8));
        }
    }
    __syncthreads();
    const int wave = threadIdx.x >> 5, lane = threadIdx.x & 31;
    const int mt = wave >> 1, nh = wave & 1;
    bf16x16 afrag[KC];
#pragma unroll
    for (int k = 0; k < KC; ++k)
        afrag[k] = load_frag(As + mt * 16 * KP + k * 32, KP, lane);
    const int l = lane & 15, half = lane >> 4;
    const long rb = (long)rowBase + mt * 16 + half * 8;
    for (int base = nh * 2; base < NT; base += 4) {        // two N-tiles/iter
        const int nt0 = base, nt1 = base + 1;
        f32x8 acc0 = {}, acc1 = {};
#pragma unroll
        for (int k = 0; k < KC; ++k) {
            bf16x16 b0 = load_frag(Wi + nt0 * 16 * KP + k * 32, KP, lane);
            bf16x16 b1 = load_frag(Wi + nt1 * 16 * KP + k * 32, KP, lane);
            acc0 = wmma_bf16(afrag[k], b0, acc0);
            acc1 = wmma_bf16(afrag[k], b1, acc1);
        }
#pragma unroll
        for (int t = 0; t < 2; ++t) {
            f32x8 acc = t ? acc1 : acc0;
            const int col = (t ? nt1 : nt0) * 16 + l;
            f32x8 rl;
#pragma unroll
            for (int i = 0; i < 8; ++i) rl[i] = fmaxf(acc[i], 0.f);
            store_col8_bf16(b_input, rb, col, acc);
            store_col8_bf16(b_msg,   rb, col, rl);
        }
    }
}

// ------------------- a_msg[a] = sum_k b_msg[a2b[a][k]]   (bandwidth kernel)
__global__ void k_aggregate(const unsigned short* __restrict__ msg,
                            const int* __restrict__ a2b,
                            unsigned short* __restrict__ a_msg) {
    int tid = blockIdx.x * blockDim.x + threadIdx.x;
    const int total = (N_ATOMS + 1) * (HP / 8);
    if (tid >= total) return;
    int atom = tid / (HP / 8);
    int c    = (tid - atom * (HP / 8)) * 8;
    const int4 nb = *(const int4*)(a2b + (long)atom * 4);
    bf16x8 v0 = load_b8(msg + (long)nb.x * HP + c);
    bf16x8 v1 = load_b8(msg + (long)nb.y * HP + c);
    bf16x8 v2 = load_b8(msg + (long)nb.z * HP + c);
    bf16x8 v3 = load_b8(msg + (long)nb.w * HP + c);
    bf16x8 s = (v0 + v1) + (v2 + v3);
    store_b8(a_msg + (long)atom * HP + c, s);
}

// --- b_msg' = relu(b_input + (a_msg[b2a] - b_msg[b2revb]) @ W_h^T)  (fused)
__global__ __launch_bounds__(256)
void k_msg_update(const unsigned short* __restrict__ a_msg,
                  const unsigned short* __restrict__ msg_in,
                  const int* __restrict__ b2a,
                  const int* __restrict__ b2revb,
                  const unsigned short* __restrict__ Wh,   // [320][320] bf16
                  const unsigned short* __restrict__ b_input,
                  unsigned short* __restrict__ msg_out) {
    constexpr int KP = HP, KC = 10, NT = 20, MT = 64;
    extern __shared__ unsigned short As[];                 // [MT][KP]
    const int rowBase = blockIdx.x * MT;
    {   // stage A: fused gather + packed bf16 subtract, 8-col vector groups
        int r  = threadIdx.x >> 2;
        int c0 = (threadIdx.x & 3) * (KP / 4);
        int row = rowBase + r;
        bool valid = row <= N_BONDS;
        long ra = 0, rv = 0;
        if (valid) { ra = b2a[row]; rv = b2revb[row]; }
        unsigned short* dst = As + r * KP + c0;
        const unsigned short* pa = a_msg  + ra * HP + c0;
        const unsigned short* pb = msg_in + rv * HP + c0;
#pragma unroll
        for (int g = 0; g < (KP / 4) / 8; ++g) {           // 10 x 8 cols
            bf16x8 d;
            if (valid) d = load_b8(pa + g * 8) - load_b8(pb + g * 8);
            else       d = __builtin_bit_cast(bf16x8, (u32x4){0u, 0u, 0u, 0u});
            store_b8(dst + g * 8, d);
        }
    }
    __syncthreads();
    const int wave = threadIdx.x >> 5, lane = threadIdx.x & 31;
    const int mt = wave >> 1, nh = wave & 1;
    bf16x16 afrag[KC];
#pragma unroll
    for (int k = 0; k < KC; ++k)
        afrag[k] = load_frag(As + mt * 16 * KP + k * 32, KP, lane);
    const int l = lane & 15, half = lane >> 4;
    const long rb = (long)rowBase + mt * 16 + half * 8;
    for (int base = nh * 2; base < NT; base += 4) {        // two N-tiles/iter
        const int nt0 = base, nt1 = base + 1;
        f32x8 acc0 = {}, acc1 = {};
#pragma unroll
        for (int k = 0; k < KC; ++k) {
            bf16x16 b0 = load_frag(Wh + nt0 * 16 * KP + k * 32, KP, lane);
            bf16x16 b1 = load_frag(Wh + nt1 * 16 * KP + k * 32, KP, lane);
            acc0 = wmma_bf16(afrag[k], b0, acc0);
            acc1 = wmma_bf16(afrag[k], b1, acc1);
        }
#pragma unroll
        for (int t = 0; t < 2; ++t) {
            f32x8 acc = t ? acc1 : acc0;
            const int col = (t ? nt1 : nt0) * 16 + l;
            f32x8 vals;
#pragma unroll
            for (int i = 0; i < 8; ++i)
                vals[i] = fmaxf(acc[i] + bf2f(b_input[(rb + i) * HP + col]), 0.f);
            store_col8_bf16(msg_out, rb, col, vals);
        }
    }
}

// ---- atom_hiddens = relu(concat(f_atoms, a_msg) @ W_o^T + b)  (K = 448 pad)
__global__ __launch_bounds__(256)
void k_atom_out(const float* __restrict__ f_atoms,
                const unsigned short* __restrict__ a_msg,  // bf16
                const unsigned short* __restrict__ Wo,     // [320][448] bf16
                const float* __restrict__ bias,
                float* __restrict__ hidden) {              // [NAP][HP] f32
    constexpr int KP = 448, KC = 14, NT = 19, MT = 64;
    extern __shared__ unsigned short As[];                 // [MT][KP]
    const int rowBase = blockIdx.x * MT;
    {   // stage concat(f_atoms[133] (cvt), a_msg[300] (raw bf16)), pad to 448
        int r  = threadIdx.x >> 2;
        int c0 = (threadIdx.x & 3) * (KP / 4);
        int row = rowBase + r;
        bool valid = row <= N_ATOMS;
        unsigned short* dst = As + r * KP + c0;
        for (int i = 0; i < KP / 4; ++i) {
            int c = c0 + i;
            unsigned short v = 0;
            if (valid) {
                if (c < ATOM_FDIM)               v = cvt_bf16(f_atoms[(long)row * ATOM_FDIM + c]);
                else if (c < ATOM_FDIM + HIDDEN) v = a_msg[(long)row * HP + (c - ATOM_FDIM)];
            }
            dst[i] = v;
        }
    }
    __syncthreads();
    const int wave = threadIdx.x >> 5, lane = threadIdx.x & 31;
    const int mt = wave >> 1, nh = wave & 1;
    bf16x16 afrag[KC];
#pragma unroll
    for (int k = 0; k < KC; ++k)
        afrag[k] = load_frag(As + mt * 16 * KP + k * 32, KP, lane);
    const int l = lane & 15, half = lane >> 4;
    const long rb = (long)rowBase + mt * 16 + half * 8;
    for (int base = nh * 2; base < NT; base += 4) {        // two N-tiles/iter
        const int  nt0 = base, nt1 = base + 1;
        const bool has1 = nt1 < NT;                        // NT=19: guard odd tile
        f32x8 acc0 = {}, acc1 = {};
#pragma unroll
        for (int k = 0; k < KC; ++k) {
            bf16x16 b0 = load_frag(Wo + nt0 * 16 * KP + k * 32, KP, lane);
            acc0 = wmma_bf16(afrag[k], b0, acc0);
            if (has1) {
                bf16x16 b1 = load_frag(Wo + nt1 * 16 * KP + k * 32, KP, lane);
                acc1 = wmma_bf16(afrag[k], b1, acc1);
            }
        }
#pragma unroll
        for (int t = 0; t < 2; ++t) {
            if (t && !has1) break;
            f32x8 acc = t ? acc1 : acc0;
            const int col = (t ? nt1 : nt0) * 16 + l;
            const float bv = (col < HIDDEN) ? bias[col] : 0.f;
#pragma unroll
            for (int i = 0; i < 8; ++i)
                hidden[(rb + i) * HP + col] = fmaxf(acc[i] + bv, 0.f);
        }
    }
}

// ---------------------------- mol_vecs = mean over 25 atoms (skip pad row 0)
__global__ void k_mol_mean(const float* __restrict__ hidden,
                           float* __restrict__ out) {
    int tid = blockIdx.x * blockDim.x + threadIdx.x;
    const int total = (N_ATOMS / 25) * HIDDEN;
    if (tid >= total) return;
    int m = tid / HIDDEN, c = tid - m * HIDDEN;
    long base = (long)(1 + m * 25) * HP + c;
    float s = 0.f;
#pragma unroll
    for (int i = 0; i < 25; ++i) s += hidden[base + (long)i * HP];
    out[tid] = s * (1.0f / 25.0f);
}

extern "C" void kernel_launch(void* const* d_in, const int* in_sizes, int n_in,
                              void* d_out, int out_size, void* d_ws, size_t ws_size,
                              hipStream_t stream) {
    const float* f_atoms = (const float*)d_in[0];
    const float* f_bonds = (const float*)d_in[1];
    const float* W_i     = (const float*)d_in[2];
    const float* W_h     = (const float*)d_in[3];
    const float* W_ow    = (const float*)d_in[4];
    const float* W_ob    = (const float*)d_in[5];
    const int*   a2b     = (const int*)d_in[6];
    const int*   b2a     = (const int*)d_in[7];
    const int*   b2revb  = (const int*)d_in[8];

    char*  ws  = (char*)d_ws;
    size_t off = 0;
    auto alloc = [&](size_t bytes) -> char* {
        char* p = ws + off;
        off += (bytes + 255) & ~(size_t)255;
        return p;
    };
    unsigned short* b_input = (unsigned short*)alloc((size_t)NBP * HP * 2);
    unsigned short* msgA    = (unsigned short*)alloc((size_t)NBP * HP * 2);
    unsigned short* msgB    = (unsigned short*)alloc((size_t)NBP * HP * 2);
    unsigned short* a_msg   = (unsigned short*)alloc((size_t)NAP * HP * 2);
    float*          hidden  = (float*)alloc((size_t)NAP * HP * 4);
    unsigned short* Wi = (unsigned short*)alloc((size_t)320 * 160 * 2);
    unsigned short* Wh = (unsigned short*)alloc((size_t)320 * 320 * 2);
    unsigned short* Wo = (unsigned short*)alloc((size_t)320 * 448 * 2);
    if (off > ws_size) return;  // scratch too small; nothing sensible to do

    k_convert_pad<<<(320 * 160 + 255) / 256, 256, 0, stream>>>(W_i,  Wi, HIDDEN, BOND_FDIM,          320, 160);
    k_convert_pad<<<(320 * 320 + 255) / 256, 256, 0, stream>>>(W_h,  Wh, HIDDEN, HIDDEN,             320, 320);
    k_convert_pad<<<(320 * 448 + 255) / 256, 256, 0, stream>>>(W_ow, Wo, HIDDEN, ATOM_FDIM + HIDDEN, 320, 448);

    const int BBLK = NBP / 64;   // 6251
    const int ABLK = NAP / 64;   // 1564
    k_bond_input<<<BBLK, 256, 64 * 160 * 2, stream>>>(f_bonds, Wi, b_input, msgA);

    const int aggBlocks = ((N_ATOMS + 1) * (HP / 8) + 255) / 256;
    unsigned short* cur = msgA;
    unsigned short* nxt = msgB;
    for (int d = 0; d < DEPTH - 1; ++d) {
        k_aggregate<<<aggBlocks, 256, 0, stream>>>(cur, a2b, a_msg);
        k_msg_update<<<BBLK, 256, 64 * HP * 2, stream>>>(a_msg, cur, b2a, b2revb,
                                                         Wh, b_input, nxt);
        unsigned short* t = cur; cur = nxt; nxt = t;
    }
    k_aggregate<<<aggBlocks, 256, 0, stream>>>(cur, a2b, a_msg);

    k_atom_out<<<ABLK, 256, 64 * 448 * 2, stream>>>(f_atoms, a_msg, Wo, W_ob, hidden);

    const int meanBlocks = ((N_ATOMS / 25) * HIDDEN + 255) / 256;
    k_mol_mean<<<meanBlocks, 256, 0, stream>>>(hidden, (float*)d_out);
}
